// LSTM_36893769073188
// MI455X (gfx1250) — compile-verified, matrix operand order checked by reading, not run on hardware
//
#include <hip/hip_runtime.h>

// ---------------- problem constants ----------------
#define DIM  1024          // D (input feature dim)
#define HDIM 1024          // H (hidden dim)
#define NB   64            // batch N
#define TT   512           // timesteps T
#define K2   (DIM + HDIM)  // fused K = 2048
#define NKCH (K2 / 32)     // 64 K-chunks of 32
// LDS layout: Bs[col][kchunk][gate][32] + 8-element pad per column.
// col stride = 64*4*32 + 8 = 8200 elems = 16400 B (16B-aligned, breaks the
// 256B bank cycle so lanes 0..15 hit distinct banks on ds_load_b128).
#define COLSTRIDE (NKCH * 4 * 32 + 8)

typedef __bf16 bf16_t;
typedef __attribute__((ext_vector_type(16))) __bf16 v16bf;
typedef __attribute__((ext_vector_type(8)))  __bf16 v8bf;
typedef __attribute__((ext_vector_type(8)))  float  v8f;

__device__ __forceinline__ float sigf(float x) {
  return 1.0f / (1.0f + __expf(-x));
}

// ---------------- prep kernels ----------------

// fp32 x -> bf16 x  (N*T*D elements)
__global__ void cvt_x_kernel(const float* __restrict__ x,
                             bf16_t* __restrict__ xb, int n) {
  int i = blockIdx.x * blockDim.x + threadIdx.x;
  if (i < n) xb[i] = (bf16_t)x[i];
}

// Build transposed, concatenated bf16 weights:
// Wt[ncol][k] = (k < D) ? Wx[k][ncol] : Wh[k-D][ncol],  shape (4H, K2)
__global__ void build_wt_kernel(const float* __restrict__ Wx,
                                const float* __restrict__ Wh,
                                bf16_t* __restrict__ wt) {
  int i = blockIdx.x * blockDim.x + threadIdx.x;      // 4H * K2 = 8388608
  int k    = i & (K2 - 1);
  int ncol = i >> 11;
  float v = (k < DIM) ? Wx[(size_t)k * (4 * HDIM) + ncol]
                      : Wh[(size_t)(k - DIM) * (4 * HDIM) + ncol];
  wt[i] = (bf16_t)v;
}

// h0 fp32 -> bf16 ping buffer, zero the grid-barrier state
__global__ void init_kernel(const float* __restrict__ h0,
                            bf16_t* __restrict__ hbuf0,
                            unsigned* __restrict__ bar) {
  int i = blockIdx.x * blockDim.x + threadIdx.x;
  if (i < NB * HDIM) hbuf0[i] = (bf16_t)h0[i];
  if (i < 2) bar[i] = 0u;
}

// ---------------- device-wide barrier (sense-reversing) ----------------
__device__ __forceinline__ void grid_barrier(unsigned* bar, unsigned nblocks) {
  __threadfence();          // release all this thread's global stores to L2
  __syncthreads();
  if (threadIdx.x == 0) {
    unsigned gen = __hip_atomic_load(&bar[1], __ATOMIC_RELAXED,
                                     __HIP_MEMORY_SCOPE_AGENT);
    unsigned arrived = __hip_atomic_fetch_add(&bar[0], 1u, __ATOMIC_ACQ_REL,
                                              __HIP_MEMORY_SCOPE_AGENT) + 1u;
    if (arrived == nblocks) {
      __hip_atomic_store(&bar[0], 0u, __ATOMIC_RELAXED,
                         __HIP_MEMORY_SCOPE_AGENT);
      __hip_atomic_fetch_add(&bar[1], 1u, __ATOMIC_RELEASE,
                             __HIP_MEMORY_SCOPE_AGENT);
    } else {
      while (__hip_atomic_load(&bar[1], __ATOMIC_RELAXED,
                               __HIP_MEMORY_SCOPE_AGENT) == gen) {
        __builtin_amdgcn_s_sleep(2);
      }
    }
    __threadfence();        // acquire: invalidate stale lines before reads
  }
  __syncthreads();
}

// ---------------- persistent fused LSTM kernel ----------------
// grid  = 64 blocks : blockIdx.x = h-tile (16 hidden columns)
// block = 128 thr   : wave w (0..3) = m-tile (16 batch rows)
__global__ void __launch_bounds__(128, 1)
lstm_persistent_kernel(const bf16_t* __restrict__ xb,   // (N,T,D) bf16
                       const bf16_t* __restrict__ wt,   // (4H,K2) bf16
                       const float*  __restrict__ bias, // (4H,)  fp32
                       bf16_t* __restrict__ hbuf0,      // (N,H) bf16 ping
                       bf16_t* __restrict__ hbuf1,      // (N,H) bf16 pong
                       float*  __restrict__ out,        // (N,T,H) fp32
                       unsigned* __restrict__ bar) {
  // Gate-interleaved weight tile, resident for all 512 steps (~256 KB of 320 KB)
  __shared__ __align__(16) bf16_t Bs[16][COLSTRIDE];

  const int tid   = threadIdx.x;
  const int lane  = tid & 31;
  const int mtile = tid >> 5;       // 0..3
  const int htile = blockIdx.x;     // 0..63

  // ---- stage weights into LDS once, gate-interleaved:
  // dst elem (within col cc) = (k>>5)*128 + g*32 + (k&31)
  for (int idx = tid; idx < 4 * 16 * (K2 / 8); idx += 128) {
    int k8 = idx & (K2 / 8 - 1);
    int cc = (idx >> 8) & 15;
    int g  = idx >> 12;
    int k  = k8 * 8;
    v8bf v = *(const v8bf*)(wt + (size_t)(g * HDIM + htile * 16 + cc) * K2 + k);
    *(v8bf*)(&Bs[cc][((k >> 5) * 128) + g * 32 + (k & 31)]) = v;
  }
  __syncthreads();

  // ---- per-lane constants (CDNA5 16-bit WMMA fragment layout)
  const int colN = lane & 15;                       // local col
  const int gcol = htile * 16 + colN;               // global hidden col
  const int rowm = mtile * 16 + (lane & 15);        // A-fragment row
  const int koff = (lane < 16) ? 0 : 8;             // half-wave K offset
  const int mbase = mtile * 16 + ((lane < 16) ? 0 : 8); // C/D row base

  // Per-lane LDS base: all 8 B-fragment loads of a K-step use immediates
  // {0,32,64,...,224} bytes off (bcol + kchunk*128 elems).
  const bf16_t* bcol = &Bs[colN][koff];

  const float bI = bias[gcol];
  const float bF = bias[HDIM + gcol];
  const float bO = bias[2 * HDIM + gcol];
  const float bG = bias[3 * HDIM + gcol];

  float creg[8];                                    // cell state in VGPRs
#pragma unroll
  for (int r = 0; r < 8; ++r) creg[r] = 0.0f;

  const bf16_t* xrow_base = xb + (size_t)rowm * TT * DIM;

  for (int t = 0; t < TT; ++t) {
    const bf16_t* hcur = (t & 1) ? hbuf1 : hbuf0;
    bf16_t*       hnxt = (t & 1) ? hbuf0 : hbuf1;

    v8f accI = {}; v8f accF = {}; v8f accO = {}; v8f accG = {};

    const bf16_t* xrow = xrow_base + (size_t)t * DIM;
    const bf16_t* hrow = hcur + rowm * HDIM;

#define KSTEP(APTR, KC)                                                       \
    {                                                                         \
      const bf16_t* bp = bcol + (KC) * 128;                                   \
      union { v16bf v; v8bf h[2]; } af, b0, b1, b2, b3;                       \
      af.h[0] = *(const v8bf*)(APTR);                                         \
      af.h[1] = *(const v8bf*)((APTR) + 16);                                  \
      b0.h[0] = *(const v8bf*)(bp);                                           \
      b0.h[1] = *(const v8bf*)(bp + 16);                                      \
      b1.h[0] = *(const v8bf*)(bp + 32);                                      \
      b1.h[1] = *(const v8bf*)(bp + 48);                                      \
      b2.h[0] = *(const v8bf*)(bp + 64);                                      \
      b2.h[1] = *(const v8bf*)(bp + 80);                                      \
      b3.h[0] = *(const v8bf*)(bp + 96);                                      \
      b3.h[1] = *(const v8bf*)(bp + 112);                                     \
      accI = __builtin_amdgcn_wmma_f32_16x16x32_bf16(false, af.v, false,      \
                 b0.v, (short)0, accI, false, false);                         \
      accF = __builtin_amdgcn_wmma_f32_16x16x32_bf16(false, af.v, false,      \
                 b1.v, (short)0, accF, false, false);                         \
      accO = __builtin_amdgcn_wmma_f32_16x16x32_bf16(false, af.v, false,      \
                 b2.v, (short)0, accO, false, false);                         \
      accG = __builtin_amdgcn_wmma_f32_16x16x32_bf16(false, af.v, false,      \
                 b3.v, (short)0, accG, false, false);                         \
    }

    // K = [0, D): contribution of x_t   (kchunk = k0/32)
#pragma unroll 8
    for (int k0 = 0; k0 < DIM; k0 += 32) {
      KSTEP(xrow + k0 + koff, (k0 >> 5))
    }
    // K = [D, D+H): contribution of h_{t-1}  (kchunk = 32 + k0/32)
#pragma unroll 8
    for (int k0 = 0; k0 < HDIM; k0 += 32) {
      KSTEP(hrow + k0 + koff, (k0 >> 5) + (DIM / 32))
    }
#undef KSTEP

    // ---- gates + state update (C/D layout: elem r, lane -> M=mbase+r, N=colN)
#pragma unroll
    for (int r = 0; r < 8; ++r) {
      int m = mbase + r;
      float vi = sigf(accI[r] + bI);
      float vf = sigf(accF[r] + bF);
      float vo = sigf(accO[r] + bO);
      float vg = tanhf(accG[r] + bG);
      float cn = vf * creg[r] + vi * vg;
      creg[r] = cn;
      float hn = vo * tanhf(cn);
      out[((size_t)m * TT + t) * HDIM + gcol] = hn;
      hnxt[(size_t)m * HDIM + gcol] = (bf16_t)hn;
    }

    grid_barrier(bar, gridDim.x);   // publish h_t before step t+1
  }
}

// ---------------- host launcher ----------------
extern "C" void kernel_launch(void* const* d_in, const int* in_sizes, int n_in,
                              void* d_out, int out_size, void* d_ws, size_t ws_size,
                              hipStream_t stream) {
  const float* x  = (const float*)d_in[0];   // (N,T,D)
  const float* h0 = (const float*)d_in[1];   // (N,H)
  const float* Wx = (const float*)d_in[2];   // (D,4H)
  const float* Wh = (const float*)d_in[3];   // (H,4H)
  const float* b  = (const float*)d_in[4];   // (4H,)
  float* out = (float*)d_out;                // (N,T,H)

  // workspace carve-up (all offsets 256B aligned)
  char* ws = (char*)d_ws;
  size_t off = 0;
  unsigned* bar = (unsigned*)(ws + off);            off += 256;
  bf16_t* xb    = (bf16_t*)(ws + off);              off += (size_t)NB * TT * DIM * 2;  // 64 MiB
  bf16_t* wt    = (bf16_t*)(ws + off);              off += (size_t)4 * HDIM * K2 * 2;  // 16 MiB
  bf16_t* hb0   = (bf16_t*)(ws + off);              off += (size_t)NB * HDIM * 2;
  bf16_t* hb1   = (bf16_t*)(ws + off);              off += (size_t)NB * HDIM * 2;
  (void)ws_size; (void)in_sizes; (void)n_in; (void)out_size;

  // 1) x -> bf16
  {
    int n = NB * TT * DIM;
    cvt_x_kernel<<<(n + 255) / 256, 256, 0, stream>>>(x, xb, n);
  }
  // 2) transpose+concat weights -> bf16
  {
    int n = 4 * HDIM * K2;
    build_wt_kernel<<<(n + 255) / 256, 256, 0, stream>>>(Wx, Wh, wt);
  }
  // 3) h0 -> bf16, zero barrier
  {
    int n = NB * HDIM;
    init_kernel<<<(n + 255) / 256, 256, 0, stream>>>(h0, hb0, bar);
  }
  // 4) persistent fused recurrence: 64 blocks (h-tiles) x 128 threads (4 waves)
  lstm_persistent_kernel<<<64, 128, 0, stream>>>(xb, wt, b, hb0, hb1, out, bar);
}
